// HyenaOperator_32796370272336
// MI455X (gfx1250) — matrix operand order repaired and checked by reading, not compiled, not run on hardware
//
#include <hip/hip_runtime.h>
#include <hip/hip_bf16.h>
#include <math.h>

// ---------------------------------------------------------------- constants
#define Bsz   4
#define Lseq  8192
#define Dm    768
#define D3    (3 * Dm)
#define FO    64
#define N2    16384        // FFT length (2*L)
#define LOG2N 14

typedef __bf16 v16bf __attribute__((ext_vector_type(16)));
typedef __bf16 v8bf  __attribute__((ext_vector_type(8)));
typedef float  v8f   __attribute__((ext_vector_type(8)));

// exact pointee type the async builtin expects: GCC-style int vector, 16 bytes
typedef int v4i_b128 __attribute__((vector_size(16)));
typedef __attribute__((address_space(1))) v4i_b128* gp_b128;
typedef __attribute__((address_space(3))) v4i_b128* lp_b128;

// ---------------------------------------------------------------- async global->LDS (CDNA5), guarded
#if defined(__has_builtin)
#  if __has_builtin(__builtin_amdgcn_global_load_async_to_lds_b128)
#    define ASYNC_OK 1
#  endif
#endif
#ifndef ASYNC_OK
#  define ASYNC_OK 0
#endif

// copy 16 bytes global -> LDS (async if available, else sync through VGPRs)
static __device__ __forceinline__ void async_b128(const void* gsrc, void* ldst) {
#if ASYNC_OK
    gp_b128 g = (gp_b128)(unsigned long long)gsrc;
    lp_b128 l = (lp_b128)(unsigned int)(unsigned long long)ldst;
    __builtin_amdgcn_global_load_async_to_lds_b128(g, l, 0, 0);
#else
    *(uint4*)ldst = *(const uint4*)gsrc;
#endif
}

static __device__ __forceinline__ void async_wait0() {
#if ASYNC_OK
#  if __has_builtin(__builtin_amdgcn_s_wait_asynccnt)
    __builtin_amdgcn_s_wait_asynccnt(0);
#  else
    asm volatile("s_wait_asynccnt 0x0" ::: "memory");
#  endif
#endif
}

// ---------------------------------------------------------------- helpers
static __device__ __forceinline__ unsigned short f32_to_bf16(float f) {
    union { float f; unsigned int u; } v; v.f = f;
    unsigned int r = v.u + 0x7FFFu + ((v.u >> 16) & 1u);   // round-nearest-even
    return (unsigned short)(r >> 16);
}

// ---------------------------------------------------------------- cvt f32 -> bf16
__global__ void cvt_bf16_kernel(const float* __restrict__ in,
                                unsigned short* __restrict__ out, int n) {
    int i = blockIdx.x * blockDim.x + threadIdx.x;
    if (i < n) out[i] = f32_to_bf16(in[i]);
}

// ---------------------------------------------------------------- pack W (KxN, f32) into WMMA-B fragments (bf16)
// layout: Bp[((kb*(N/16)+nb)*512) + lane*16 + i]
//   value = W[(kb*32 + (i<8 ? (lane/16)*8 + i : 16 + (lane/16)*8 + i-8)) * N + nb*16 + (lane%16)]
__global__ void pack_b_kernel(const float* __restrict__ W,
                              unsigned short* __restrict__ Bp, int K, int N) {
    int idx = blockIdx.x * blockDim.x + threadIdx.x;
    int total = K * N;
    if (idx >= total) return;
    int i    = idx & 15;
    int lane = (idx >> 4) & 31;
    int tile = idx >> 9;
    int ntiles = N >> 4;
    int nb = tile % ntiles;
    int kb = tile / ntiles;
    int k = kb * 32 + (i < 8 ? (lane >> 4) * 8 + i : 16 + (lane >> 4) * 8 + (i - 8));
    int n = nb * 16 + (lane & 15);
    Bp[idx] = f32_to_bf16(W[(size_t)k * N + n]);
}

// ---------------------------------------------------------------- WMMA bf16 GEMM  C = A(bf16,MxK) * Bpacked + bias
// block = 128 threads (4 wave32); block tile 64(M) x 64(N); each wave 16x64.
// B tile (4KB, shared by all 4 waves) double-buffered in LDS via async copies.
// A fragments software-pipelined through registers; B fragments batched so a
// single DS wait covers the whole 4-WMMA chain.
__global__ __launch_bounds__(128)
void wmma_gemm_kernel(const unsigned short* __restrict__ A,
                      const unsigned short* __restrict__ Bp,
                      const float* __restrict__ bias,
                      float* __restrict__ C, int M, int K, int N) {
    __shared__ __align__(32) unsigned short sB[2][2048];   // 2 x 4KB

    const int tid  = threadIdx.x;
    const int lane = tid & 31;
    const int wave = tid >> 5;
    const int half = lane >> 4;
    const int l16  = lane & 15;
    const int m0    = blockIdx.y * 64 + wave * 16;
    const int nBase = blockIdx.x * 64;
    const int ntiles = N >> 4;
    const int nb0 = nBase >> 4;
    const int kblocks = K >> 5;

    v8f acc[4] = {};

    const int row = m0 + l16;
    const unsigned short* arow = A + (size_t)row * K + half * 8;

    // issue async copy of B tile kb (2048 bf16, contiguous) into sB[buf]
    auto loadB = [&](int kb, int buf) {
        const unsigned short* src = Bp + ((size_t)kb * ntiles + nb0) * 512 + tid * 16;
        unsigned short* dst = &sB[buf][tid * 16];
        async_b128(src, dst);
        async_b128(src + 8, dst + 8);
    };

    loadB(0, 0);

    // preload A fragment for kb = 0
    v8bf alo = *(const v8bf*)(arow);
    v8bf ahi = *(const v8bf*)(arow + 16);

    async_wait0();
    __syncthreads();

    for (int kb = 0; kb < kblocks; ++kb) {
        const int buf = kb & 1;
        if (kb + 1 < kblocks) loadB(kb + 1, buf ^ 1);   // async prefetch next B tile

        v16bf a = __builtin_shufflevector(alo, ahi,
                   0,1,2,3,4,5,6,7,8,9,10,11,12,13,14,15);

        // issue next A fragment loads early so they overlap the WMMA chain
        if (kb + 1 < kblocks) {
            const unsigned short* ap = arow + (kb + 1) * 32;
            __builtin_prefetch(ap + 32, 0, 0);
            alo = *(const v8bf*)(ap);
            ahi = *(const v8bf*)(ap + 16);
        }

        // batch all 4 B-fragment DS loads (single wait for the chain)
        v16bf bfrag[4];
#pragma unroll
        for (int nt = 0; nt < 4; ++nt)
            bfrag[nt] = *(const v16bf*)(&sB[buf][nt * 512 + lane * 16]);

#pragma unroll
        for (int nt = 0; nt < 4; ++nt)
            acc[nt] = __builtin_amdgcn_wmma_f32_16x16x32_bf16(
                false, a, false, bfrag[nt], (short)0, acc[nt], false, false);

        async_wait0();          // next tile landed
        __syncthreads();        // everyone done reading buf before it is reused
    }

#pragma unroll
    for (int nt = 0; nt < 4; ++nt) {
        int col = nBase + nt * 16 + l16;
        float bv = bias[col];
#pragma unroll
        for (int r = 0; r < 8; ++r) {
            int rrow = m0 + half * 8 + r;
            C[(size_t)rrow * N + col] = acc[nt][r] + bv;
        }
    }
}

// ---------------------------------------------------------------- depthwise causal conv(3) + split + gate v*x2
// z: [B, L, 3D] f32 ; outputs channel-major [B, D, L]
__global__ void conv_split_kernel(const float* __restrict__ z,
                                  const float* __restrict__ cw,   // [3,3D]
                                  const float* __restrict__ cb,   // [3D]
                                  float* __restrict__ x1_ch,
                                  float* __restrict__ v_ch) {
    int idx = blockIdx.x * blockDim.x + threadIdx.x;
    const int total = Bsz * Dm * Lseq;
    if (idx >= total) return;
    int t = idx % Lseq;
    int d = (idx / Lseq) % Dm;
    int b = idx / (Lseq * Dm);

    float vals[3];
#pragma unroll
    for (int p = 0; p < 3; ++p) {
        int c = p * Dm + d;
        float acc = cb[c];
        size_t base = ((size_t)b * Lseq) * D3 + c;
#pragma unroll
        for (int tap = 0; tap < 3; ++tap) {
            int tt = t - 2 + tap;
            if (tt >= 0)
                acc += z[base + (size_t)tt * D3] * cw[tap * D3 + c];
        }
        vals[p] = acc;
    }
    size_t o = ((size_t)b * Dm + d) * Lseq + t;
    x1_ch[o] = vals[0];
    v_ch[o]  = vals[2] * vals[1];      // v * x2
}

// ---------------------------------------------------------------- filter MLP: h2[L,64]
__global__ __launch_bounds__(64)
void hyena_mlp_kernel(const float* __restrict__ freq,
                      const float* __restrict__ W0, const float* __restrict__ b0,
                      const float* __restrict__ W1, const float* __restrict__ b1,
                      const float* __restrict__ W2, const float* __restrict__ b2,
                      float* __restrict__ h2) {
    const int t = blockIdx.x;
    const int o = threadIdx.x;
    __shared__ float sa[FO];
    __shared__ float sb[FO];

    float tt   = (float)t * (1.0f / (float)(Lseq - 1));
    float wang = 6.28318530717958647692f * (float)t / (float)Lseq;
    float z0 = tt;
    float z1 =  __cosf(1e-4f * wang);
    float z2 = -__sinf(1e-4f * wang);

    float fq = freq[o];
    float a = z0 * W0[o] + z1 * W0[FO + o] + z2 * W0[2 * FO + o] + b0[o];
    sa[o] = __sinf(fq * a);
    __syncthreads();

    float acc = b1[o];
#pragma unroll 8
    for (int i = 0; i < FO; ++i) acc += sa[i] * W1[i * FO + o];
    sb[o] = __sinf(fq * acc);
    __syncthreads();

    acc = b2[o];
#pragma unroll 8
    for (int i = 0; i < FO; ++i) acc += sb[i] * W2[i * FO + o];
    h2[(size_t)t * FO + o] = __sinf(fq * acc);
}

// ---------------------------------------------------------------- k_ch[d][t] = (h2[t]·Wf[:,d]) * exp(-t_norm*|delta_d|)
__global__ void filter_final_kernel(const float* __restrict__ h2,
                                    const float* __restrict__ Wf,
                                    float* __restrict__ k_ch) {
    int idx = blockIdx.x * blockDim.x + threadIdx.x;
    const int total = Dm * Lseq;
    if (idx >= total) return;
    int t = idx % Lseq;
    int d = idx / Lseq;
    float acc = 0.f;
    const float* hrow = h2 + (size_t)t * FO;
#pragma unroll 8
    for (int i = 0; i < FO; ++i) acc += hrow[i] * Wf[i * Dm + d];
    const float ln001 = -4.60517018598809136804f;           // log(0.01)
    float mind = ln001 / 1.5f;                               // -3.0701
    float maxd = ln001 / 0.3f;                               // -15.3506
    float delta = mind + (maxd - mind) * (float)d / (float)(Dm - 1);
    float tt = (float)t * (1.0f / (float)(Lseq - 1));
    k_ch[(size_t)d * Lseq + t] = acc * __expf(-tt * fabsf(delta));
}

// ---------------------------------------------------------------- in-LDS radix-2 iterative FFT (size N2)
static __device__ void fft_lds(float* re, float* im, float sgn) {
    const int tid  = threadIdx.x;
    const int nthr = blockDim.x;
    // bit reversal
    for (int i = tid; i < N2; i += nthr) {
        int j = (int)(__brev((unsigned)i) >> (32 - LOG2N));
        if (j > i) {
            float tr = re[i]; re[i] = re[j]; re[j] = tr;
            float ti = im[i]; im[i] = im[j]; im[j] = ti;
        }
    }
    __syncthreads();
    for (int s = 1; s <= LOG2N; ++s) {
        const int m    = 1 << s;
        const int half = m >> 1;
        const float ang0 = sgn * 6.28318530717958647692f / (float)m;
        for (int bfy = tid; bfy < (N2 >> 1); bfy += nthr) {
            int k    = bfy & (half - 1);
            int base = ((bfy >> (s - 1)) << s) + k;
            float ang = ang0 * (float)k;
            float c = __cosf(ang);
            float sn = __sinf(ang);
            float xr = re[base + half], xi = im[base + half];
            float tr = xr * c - xi * sn;
            float ti = xr * sn + xi * c;
            re[base + half] = re[base] - tr;
            im[base + half] = im[base] - ti;
            re[base] += tr;
            im[base] += ti;
        }
        __syncthreads();
    }
}

// stage a length-Lseq f32 sequence into re[] (async), zero-pad + zero im[]
static __device__ __forceinline__ void fft_stage_load(const float* __restrict__ src,
                                                      float* re, float* im) {
    const int tid = threadIdx.x, nthr = blockDim.x;
    for (int c = tid; c < (Lseq / 4); c += nthr)          // 16B chunks of payload
        async_b128(src + 4 * c, re + 4 * c);
    for (int i = Lseq + tid; i < N2; i += nthr) re[i] = 0.f;
    for (int i = tid; i < N2; i += nthr) im[i] = 0.f;
    async_wait0();
    __syncthreads();
}

// ---------------------------------------------------------------- kernel spectrum: Kf[d][j] = DFT(k_d)[j] / N2
__global__ void kf_kernel(const float* __restrict__ k_ch, float2* __restrict__ Kf) {
    extern __shared__ float smem[];
    float* re = smem;
    float* im = smem + N2;
    const int d = blockIdx.x;
    const int tid = threadIdx.x, nthr = blockDim.x;
    fft_stage_load(k_ch + (size_t)d * Lseq, re, im);
    fft_lds(re, im, -1.0f);
    const float inv = 1.0f / (float)N2;
    for (int j = tid; j < N2; j += nthr) {
        float2 o; o.x = re[j] * inv; o.y = im[j] * inv;
        Kf[(size_t)d * N2 + j] = o;
    }
}

// ---------------------------------------------------------------- FFT long conv + bias-skip + x1 gate + transpose to bf16
__global__ void fftconv_kernel(const float* __restrict__ v_ch,
                               const float* __restrict__ x1_ch,
                               const float2* __restrict__ Kf,
                               const float* __restrict__ fbias,
                               unsigned short* __restrict__ ybf) {
    extern __shared__ float smem[];
    float* re = smem;
    float* im = smem + N2;
    const int bd = blockIdx.x;
    const int b = bd / Dm;
    const int d = bd % Dm;
    const int tid = threadIdx.x, nthr = blockDim.x;
    const size_t vbase = (size_t)bd * Lseq;

    fft_stage_load(v_ch + vbase, re, im);
    fft_lds(re, im, -1.0f);                       // forward DFT
    for (int j = tid; j < N2; j += nthr) {        // pointwise spectrum multiply
        float2 kk = Kf[(size_t)d * N2 + j];
        float vr = re[j], vi = im[j];
        re[j] = vr * kk.x - vi * kk.y;
        im[j] = vr * kk.y + vi * kk.x;
    }
    __syncthreads();
    fft_lds(re, im, 1.0f);                        // inverse (1/N folded in Kf)

    const float fb = fbias[d];
    for (int t = tid; t < Lseq; t += nthr) {
        float vv = v_ch[vbase + t];
        float y = re[t] + vv * fb;
        y *= x1_ch[vbase + t];
        ybf[((size_t)b * Lseq + t) * Dm + d] = f32_to_bf16(y);
    }
}

// ---------------------------------------------------------------- workspace layout
#define SZ_UBF   ((size_t)Bsz * Lseq * Dm * 2)          // u bf16
#define SZ_WINP  ((size_t)Dm * D3 * 2)                  // packed W_in bf16
#define SZ_WOUTP ((size_t)Dm * Dm * 2)                  // packed W_out bf16
#define SZ_Z     ((size_t)Bsz * Lseq * D3 * 4)          // z f32
#define SZ_X1    ((size_t)Bsz * Dm * Lseq * 4)
#define SZ_V     ((size_t)Bsz * Dm * Lseq * 4)
#define SZ_H2    ((size_t)Lseq * FO * 4)
#define SZ_K     ((size_t)Dm * Lseq * 4)
#define SZ_KF    ((size_t)Dm * N2 * 8)
#define SZ_YBF   ((size_t)Bsz * Lseq * Dm * 2)

extern "C" void kernel_launch(void* const* d_in, const int* in_sizes, int n_in,
                              void* d_out, int out_size, void* d_ws, size_t ws_size,
                              hipStream_t stream) {
    (void)in_sizes; (void)n_in; (void)out_size; (void)ws_size;

    const float* u     = (const float*)d_in[0];
    const float* W_in  = (const float*)d_in[1];
    const float* b_in  = (const float*)d_in[2];
    const float* cw    = (const float*)d_in[3];
    const float* cb    = (const float*)d_in[4];
    const float* freq  = (const float*)d_in[5];
    const float* W0    = (const float*)d_in[6];
    const float* b0    = (const float*)d_in[7];
    const float* W1    = (const float*)d_in[8];
    const float* b1    = (const float*)d_in[9];
    const float* W2    = (const float*)d_in[10];
    const float* b2    = (const float*)d_in[11];
    const float* Wf    = (const float*)d_in[12];
    const float* fbias = (const float*)d_in[13];
    const float* W_out = (const float*)d_in[14];
    const float* b_out = (const float*)d_in[15];
    float* out = (float*)d_out;

    char* ws = (char*)d_ws;
    size_t off = 0;
    unsigned short* UBF   = (unsigned short*)(ws + off); off += SZ_UBF;
    unsigned short* WINP  = (unsigned short*)(ws + off); off += SZ_WINP;
    unsigned short* WOUTP = (unsigned short*)(ws + off); off += SZ_WOUTP;
    float*          Z     = (float*)(ws + off);          off += SZ_Z;
    float*          X1    = (float*)(ws + off);          off += SZ_X1;
    float*          V     = (float*)(ws + off);          off += SZ_V;
    float*          H2    = (float*)(ws + off);          off += SZ_H2;
    float*          KCH   = (float*)(ws + off);          off += SZ_K;
    float2*         KF    = (float2*)(ws + off);         off += SZ_KF;
    unsigned short* YBF   = (unsigned short*)(ws + off); off += SZ_YBF;

    const int M = Bsz * Lseq;                 // 32768

    // 1. u -> bf16
    {
        int n = M * Dm;
        cvt_bf16_kernel<<<(n + 255) / 256, 256, 0, stream>>>(u, UBF, n);
    }
    // 2. pack weights into WMMA-B fragment layout
    {
        int n = Dm * D3;
        pack_b_kernel<<<(n + 255) / 256, 256, 0, stream>>>(W_in, WINP, Dm, D3);
        n = Dm * Dm;
        pack_b_kernel<<<(n + 255) / 256, 256, 0, stream>>>(W_out, WOUTP, Dm, Dm);
    }
    // 3. z = u @ W_in + b_in   (WMMA, async LDS double-buffered B)
    {
        dim3 grid(D3 / 64, M / 64);
        wmma_gemm_kernel<<<grid, 128, 0, stream>>>(UBF, WINP, b_in, Z, M, Dm, D3);
    }
    // 4. depthwise conv + split + v*x2
    {
        int n = Bsz * Dm * Lseq;
        conv_split_kernel<<<(n + 255) / 256, 256, 0, stream>>>(Z, cw, cb, X1, V);
    }
    // 5. filter MLP + projection/decay
    {
        hyena_mlp_kernel<<<Lseq, FO, 0, stream>>>(freq, W0, b0, W1, b1, W2, b2, H2);
        int n = Dm * Lseq;
        filter_final_kernel<<<(n + 255) / 256, 256, 0, stream>>>(H2, Wf, KCH);
    }
    // 6. kernel spectrum (one 128KB-LDS FFT per channel)
    {
        kf_kernel<<<Dm, 256, 2 * N2 * sizeof(float), stream>>>(KCH, KF);
    }
    // 7. FFT long convolution, gate, transpose -> bf16
    {
        fftconv_kernel<<<Bsz * Dm, 256, 2 * N2 * sizeof(float), stream>>>(
            V, X1, KF, fbias, YBF);
    }
    // 8. out = y @ W_out + b_out   (WMMA)
    {
        dim3 grid(Dm / 64, M / 64);
        wmma_gemm_kernel<<<grid, 128, 0, stream>>>(YBF, WOUTP, b_out, out, M, Dm, Dm);
    }
}
